// Transformer_58368605552972
// MI455X (gfx1250) — matrix operand order repaired and checked by reading, not compiled, not run on hardware
//
#include <hip/hip_runtime.h>
#include <hip/hip_bf16.h>
#include <cstdint>
#include <cstddef>

// ---------------------------------------------------------------------------
// Types for CDNA5 WMMA (wave32): v_wmma_f32_16x16x32_bf16
// ---------------------------------------------------------------------------
typedef __attribute__((ext_vector_type(2)))  __bf16 bf16x2;
typedef __attribute__((ext_vector_type(8)))  __bf16 bf16x8;
typedef __attribute__((ext_vector_type(16))) __bf16 bf16x16;
typedef __attribute__((ext_vector_type(8)))  float  f32x8;
typedef __attribute__((ext_vector_type(4)))  unsigned int u32x4;
typedef __attribute__((ext_vector_type(8)))  int    i32x8;
typedef __attribute__((ext_vector_type(4)))  int    i32x4;

#define LDSK 40   // 32 K elements + 8 pad (16B-aligned chunks, spread banks)

#if __has_builtin(__builtin_amdgcn_tensor_load_to_lds) && \
    __has_builtin(__builtin_amdgcn_s_wait_tensorcnt)
#define USE_TDM 1
#endif

// pack two fp32 -> 2x bf16 in one dword (hardware packed convert if present)
__device__ __forceinline__ uint32_t pack_bf16_2(float a, float b) {
#if __has_builtin(__builtin_amdgcn_cvt_pk_bf16_f32)
  bf16x2 p = __builtin_amdgcn_cvt_pk_bf16_f32(a, b);
  return __builtin_bit_cast(uint32_t, p);
#else
  uint32_t ua = __builtin_bit_cast(uint32_t, a);
  ua = ua + 0x7FFFu + ((ua >> 16) & 1u);
  uint32_t ub = __builtin_bit_cast(uint32_t, b);
  ub = ub + 0x7FFFu + ((ub >> 16) & 1u);
  return (ua >> 16) | (ub & 0xFFFF0000u);
#endif
}

__device__ __forceinline__ bf16x16 frag16(const __bf16* p0, const __bf16* p1) {
  bf16x8 lo = *(const bf16x8*)p0;
  bf16x8 hi = *(const bf16x8*)p1;
  return __builtin_shufflevector(lo, hi, 0,1,2,3,4,5,6,7,8,9,10,11,12,13,14,15);
}

// ---------------------------------------------------------------------------
// Generic GEMM: C[M,N] = act( A[M,K] * W[K,N] + bias (+ residual) )
// BM=BN=128, BK=32, 256 threads = 8 waves (2x4), each wave owns 64x32.
// A-frag: lane l (m=l&15, h=l>>4) holds K [h*8,h*8+8) and [16+h*8,+8).
// B-frag: lane l (n=l&15) holds K [h*8,h*8+8) and [16+h*8,+8) from [N][K] LDS.
// D: lane holds rows h*8+e, col n.
// ---------------------------------------------------------------------------
__global__ __launch_bounds__(256)
void gemm_bf16_wmma(const float* __restrict__ A, int lda,
                    const float* __restrict__ W, int ldw,
                    const float* __restrict__ bias,
                    const float* __restrict__ res, int ldres,
                    float* __restrict__ C, int ldc,
                    int K, int relu)
{
  __shared__ __bf16 As[128][LDSK];
  __shared__ __bf16 Bs[128][LDSK];
  const int tid  = threadIdx.x;
  const int lane = tid & 31, wave = tid >> 5;
  const int wr = wave >> 2, wc = wave & 3;
  const int lm = lane & 15, lh = lane >> 4;
  const int n0 = blockIdx.x * 128;
  const int m0 = blockIdx.y * 128;

  const f32x8 z = {0.f,0.f,0.f,0.f,0.f,0.f,0.f,0.f};
  f32x8 acc[4][2];
#pragma unroll
  for (int r = 0; r < 4; ++r)
#pragma unroll
    for (int c = 0; c < 2; ++c) acc[r][c] = z;

  for (int kb = 0; kb < K; kb += 32) {
    __syncthreads();
    // A tile: 128x32 fp32 -> packed bf16 pairs
#pragma unroll
    for (int i = 0; i < 4; ++i) {
      int flat = tid + i * 256;
      int row = flat >> 3, c4 = (flat & 7) << 2;
      float4 v = *(const float4*)(A + (size_t)(m0 + row) * lda + kb + c4);
      uint32_t* d = (uint32_t*)&As[row][c4];
      d[0] = pack_bf16_2(v.x, v.y);
      d[1] = pack_bf16_2(v.z, v.w);
    }
    // B tile: W[K,N] row-major -> Bs[n][k], two k per thread -> packed store
#pragma unroll
    for (int i = 0; i < 8; ++i) {
      int flat = tid + i * 256;              // (n, kpair)
      int kp = flat >> 7, n = flat & 127;    // kp 0..15
      float w0 = W[(size_t)(kb + 2 * kp)     * ldw + n0 + n];
      float w1 = W[(size_t)(kb + 2 * kp + 1) * ldw + n0 + n];
      *(uint32_t*)&Bs[n][2 * kp] = pack_bf16_2(w0, w1);
    }
    if (kb + 32 < K) {  // hint next weight tile toward the caches
      __builtin_prefetch(W + (size_t)(kb + 32 + (tid >> 7)) * ldw + n0 + (tid & 127), 0, 1);
    }
    __syncthreads();

    bf16x16 af[4], bfr[2];
#pragma unroll
    for (int r = 0; r < 4; ++r) {
      int row = wr * 64 + r * 16 + lm;
      af[r] = frag16(&As[row][lh * 8], &As[row][16 + lh * 8]);
    }
#pragma unroll
    for (int c = 0; c < 2; ++c) {
      int col = wc * 32 + c * 16 + lm;
      bfr[c] = frag16(&Bs[col][lh * 8], &Bs[col][16 + lh * 8]);
    }
#pragma unroll
    for (int r = 0; r < 4; ++r)
#pragma unroll
      for (int c = 0; c < 2; ++c)
        acc[r][c] = __builtin_amdgcn_wmma_f32_16x16x32_bf16(
            false, af[r], false, bfr[c], (short)0, acc[r][c], false, false);
  }

#pragma unroll
  for (int r = 0; r < 4; ++r)
#pragma unroll
    for (int c = 0; c < 2; ++c) {
      int gn  = n0 + wc * 32 + c * 16 + lm;
      int gmb = m0 + wr * 64 + r * 16 + lh * 8;
      float bv = bias ? bias[gn] : 0.f;
#pragma unroll
      for (int e = 0; e < 8; ++e) {
        int gm = gmb + e;
        float v = acc[r][c][e] + bv;
        if (res)  v += res[(size_t)gm * ldres + gn];
        if (relu) v = fmaxf(v, 0.f);
        C[(size_t)gm * ldc + gn] = v;
      }
    }
}

// ---------------------------------------------------------------------------
// Attention scores: S[g][n][m] = (Q_g[n,:] . K_g[m,:]) / 8 - (1-mask)*1e4
// Q,K live in fused QKV buffer (ld 1536) at column offsets qcol/kcol + h*64.
// ---------------------------------------------------------------------------
__global__ __launch_bounds__(256)
void attn_scores_wmma(const float* __restrict__ QKV,
                      const float* __restrict__ mask,
                      float* __restrict__ S,
                      int qcol, int kcol)
{
  __shared__ __bf16 Qs[128][LDSK];
  __shared__ __bf16 Ks[128][LDSK];
  const int tid  = threadIdx.x;
  const int lane = tid & 31, wave = tid >> 5;
  const int wr = wave >> 2, wc = wave & 3;
  const int lm = lane & 15, lh = lane >> 4;
  const int n0 = blockIdx.x * 128;           // key tile
  const int m0 = blockIdx.y * 128;           // query tile
  const int g  = blockIdx.z;                 // bl*8 + h
  const int bl = g >> 3, h = g & 7;
  const float* Qb = QKV + (size_t)bl * 256 * 1536 + qcol + h * 64;
  const float* Kb = QKV + (size_t)bl * 256 * 1536 + kcol + h * 64;

  const f32x8 z = {0.f,0.f,0.f,0.f,0.f,0.f,0.f,0.f};
  f32x8 acc[4][2];
#pragma unroll
  for (int r = 0; r < 4; ++r)
#pragma unroll
    for (int c = 0; c < 2; ++c) acc[r][c] = z;

  for (int kb = 0; kb < 64; kb += 32) {
    __syncthreads();
#pragma unroll
    for (int i = 0; i < 4; ++i) {
      int flat = tid + i * 256;
      int row = flat >> 3, c4 = (flat & 7) << 2;
      float4 q = *(const float4*)(Qb + (size_t)(m0 + row) * 1536 + kb + c4);
      uint32_t* dq = (uint32_t*)&Qs[row][c4];
      dq[0] = pack_bf16_2(q.x, q.y);
      dq[1] = pack_bf16_2(q.z, q.w);
      float4 k = *(const float4*)(Kb + (size_t)(n0 + row) * 1536 + kb + c4);
      uint32_t* dk = (uint32_t*)&Ks[row][c4];
      dk[0] = pack_bf16_2(k.x, k.y);
      dk[1] = pack_bf16_2(k.z, k.w);
    }
    __syncthreads();

    bf16x16 af[4], bfr[2];
#pragma unroll
    for (int r = 0; r < 4; ++r) {
      int row = wr * 64 + r * 16 + lm;
      af[r] = frag16(&Qs[row][lh * 8], &Qs[row][16 + lh * 8]);
    }
#pragma unroll
    for (int c = 0; c < 2; ++c) {
      int col = wc * 32 + c * 16 + lm;
      bfr[c] = frag16(&Ks[col][lh * 8], &Ks[col][16 + lh * 8]);
    }
#pragma unroll
    for (int r = 0; r < 4; ++r)
#pragma unroll
      for (int c = 0; c < 2; ++c)
        acc[r][c] = __builtin_amdgcn_wmma_f32_16x16x32_bf16(
            false, af[r], false, bfr[c], (short)0, acc[r][c], false, false);
  }

  const float* mk = mask + (size_t)bl * 256 * 256;
  float* Sg = S + (size_t)g * 256 * 256;
#pragma unroll
  for (int r = 0; r < 4; ++r)
#pragma unroll
    for (int c = 0; c < 2; ++c) {
      int gn  = n0 + wc * 32 + c * 16 + lm;
      int gmb = m0 + wr * 64 + r * 16 + lh * 8;
#pragma unroll
      for (int e = 0; e < 8; ++e) {
        int gm = gmb + e;
        float v = acc[r][c][e] * 0.125f
                - (1.f - mk[(size_t)gm * 256 + gn]) * 10000.f;
        Sg[(size_t)gm * 256 + gn] = v;
      }
    }
}

// ---------------------------------------------------------------------------
// Softmax over rows of 256; emits bf16 probabilities (packed dword stores)
// 128 threads/row, 2 elements per thread.
// ---------------------------------------------------------------------------
__global__ __launch_bounds__(128)
void softmax256_bf16(const float* __restrict__ S, __bf16* __restrict__ P)
{
  __shared__ float red[128];
  const int t = threadIdx.x;
  const float* row = S + (size_t)blockIdx.x * 256;
  float v0 = row[2 * t], v1 = row[2 * t + 1];
  red[t] = fmaxf(v0, v1); __syncthreads();
  for (int s = 64; s > 0; s >>= 1) { if (t < s) red[t] = fmaxf(red[t], red[t + s]); __syncthreads(); }
  float m = red[0];
  __syncthreads();
  float e0 = __expf(v0 - m), e1 = __expf(v1 - m);
  red[t] = e0 + e1; __syncthreads();
  for (int s = 64; s > 0; s >>= 1) { if (t < s) red[t] += red[t + s]; __syncthreads(); }
  float inv = 1.f / red[0];
  ((uint32_t*)(P + (size_t)blockIdx.x * 256))[t] = pack_bf16_2(e0 * inv, e1 * inv);
}

// ---------------------------------------------------------------------------
// Attention output: O[:, h*64+d] = P_g[256x256] * V_g[256x64]
// P tiles (bf16 in global) are pulled into LDS by the Tensor Data Mover:
// 2D tile 32x128 elems, data_size=2B, TDM pad feature (+4 dwords per 16)
// reproduces the LDSK=40 row stride the WMMA fragment loads expect.
// V staged transposed (Vs[d][k]) with packed bf16 conversion.
// ---------------------------------------------------------------------------
__global__ __launch_bounds__(256)
void attn_out_wmma(const __bf16* __restrict__ P,
                   const float* __restrict__ QKV,
                   float* __restrict__ O, int vcol)
{
  __shared__ __bf16 Ps[128][LDSK];
  __shared__ __bf16 Vs[64][LDSK];
  const int tid  = threadIdx.x;
  const int lane = tid & 31, wave = tid >> 5;
  const int lm = lane & 15, lh = lane >> 4;
  const int m0 = blockIdx.x * 128;
  const int g  = blockIdx.y;
  const int bl = g >> 3, h = g & 7;
  const __bf16* Pb = P + (size_t)g * 256 * 256;
  const float*  Vb = QKV + (size_t)bl * 256 * 1536 + vcol + h * 64;

  const f32x8 z = {0.f,0.f,0.f,0.f,0.f,0.f,0.f,0.f};
  f32x8 acc[4];
#pragma unroll
  for (int c = 0; c < 4; ++c) acc[c] = z;

  for (int kb = 0; kb < 256; kb += 32) {
    __syncthreads();
#ifdef USE_TDM
    if (wave == 0) {
      // ---- Tensor DMA descriptor (D#) for the 128x32 bf16 P tile ----
      uint64_t gaddr = (uint64_t)(uintptr_t)Pb + ((size_t)m0 * 256 + kb) * 2;
      uint32_t laddr = (uint32_t)(uintptr_t)&Ps[0][0];   // LDS byte offset
      u32x4 g0;
      g0[0] = 1u;                                   // count=1, user D#
      g0[1] = laddr;                                // lds_addr
      g0[2] = (uint32_t)gaddr;                      // global_addr[31:0]
      g0[3] = (uint32_t)((gaddr >> 32) & 0x1FFFFFFu) | (2u << 30); // [56:32], type=2
      i32x8 g1;
      g1[0] = (1 << 16)     // data_size = 2 bytes
            | (1 << 20)     // pad_enable
            | (3 << 22)     // pad_interval: 16 dwords (one 32-elem row)
            | (3 << 25);    // pad_amount: 4 dwords (8 bf16) -> LDSK=40
      g1[1] = (int)(256u << 16);   // tensor_dim0 = 256 (bits 79:48, low half)
      g1[2] = (int)(256u << 16);   // tensor_dim1 = 256 (bits 111:80, low half)
      g1[3] = (int)(32u << 16);    // tile_dim0 = 32 (bits 127:112)
      g1[4] = 128;                 // tile_dim1 = 128 (bits 143:128)
      g1[5] = 256;                 // tensor_dim0_stride = 256 (bits 207:160)
      g1[6] = 0;
      g1[7] = 0;
      i32x4 zg4 = {0, 0, 0, 0};            // groups 2/3 unused (2D tensor)
      i32x8 zg8 = {0, 0, 0, 0, 0, 0, 0, 0};
      __builtin_amdgcn_tensor_load_to_lds(g0, g1, zg4, zg4, zg8, 0);
      __builtin_amdgcn_s_wait_tensorcnt(0);
    }
#else
    // manual packed copy fallback (bf16 dword pairs)
    {
      const uint32_t* Pu = (const uint32_t*)Pb;
#pragma unroll
      for (int i = 0; i < 8; ++i) {
        int flat = tid + i * 256;
        int row = flat >> 4, c2 = flat & 15;
        *(uint32_t*)&Ps[row][c2 * 2] =
            Pu[(size_t)(m0 + row) * 128 + (kb >> 1) + c2];
      }
    }
#endif
    // V tile transposed: Vs[d][k], two k per thread, packed store
#pragma unroll
    for (int i = 0; i < 4; ++i) {
      int flat = tid + i * 256;              // (dcol, kpair)
      int kp = flat >> 6, dcol = flat & 63;  // kp 0..15
      float w0 = Vb[(size_t)(kb + 2 * kp)     * 1536 + dcol];
      float w1 = Vb[(size_t)(kb + 2 * kp + 1) * 1536 + dcol];
      *(uint32_t*)&Vs[dcol][2 * kp] = pack_bf16_2(w0, w1);
    }
    __syncthreads();

    int row = wave * 16 + lm;
    bf16x16 af = frag16(&Ps[row][lh * 8], &Ps[row][16 + lh * 8]);
#pragma unroll
    for (int c = 0; c < 4; ++c) {
      int col = c * 16 + lm;
      bf16x16 bfr = frag16(&Vs[col][lh * 8], &Vs[col][16 + lh * 8]);
      acc[c] = __builtin_amdgcn_wmma_f32_16x16x32_bf16(
          false, af, false, bfr, (short)0, acc[c], false, false);
    }
  }

  const int blbase = bl * 256;
#pragma unroll
  for (int c = 0; c < 4; ++c) {
    int gn  = h * 64 + c * 16 + lm;
    int gmb = m0 + wave * 16 + lh * 8;
#pragma unroll
    for (int e = 0; e < 8; ++e)
      O[(size_t)(blbase + gmb + e) * 512 + gn] = acc[c][e];
  }
}

// ---------------------------------------------------------------------------
// LayerNorm over rows of 512 (one block per row, 2 elems/thread)
// ---------------------------------------------------------------------------
__global__ __launch_bounds__(256)
void layernorm512(const float* __restrict__ X,
                  const float* __restrict__ g,
                  const float* __restrict__ b,
                  float* __restrict__ Y)
{
  __shared__ float red[256];
  const int t = threadIdx.x;
  const float* x = X + (size_t)blockIdx.x * 512;
  float x0 = x[t], x1 = x[t + 256];
  red[t] = x0 + x1; __syncthreads();
  for (int s = 128; s > 0; s >>= 1) { if (t < s) red[t] += red[t + s]; __syncthreads(); }
  float mean = red[0] * (1.f / 512.f);
  __syncthreads();
  float d0 = x0 - mean, d1 = x1 - mean;
  red[t] = d0 * d0 + d1 * d1; __syncthreads();
  for (int s = 128; s > 0; s >>= 1) { if (t < s) red[t] += red[t + s]; __syncthreads(); }
  float inv = rsqrtf(red[0] * (1.f / 512.f) + 1e-12f);
  float* y = Y + (size_t)blockIdx.x * 512;
  y[t]       = g[t]       * d0 * inv + b[t];
  y[t + 256] = g[t + 256] * d1 * inv + b[t + 256];
}

// ---------------------------------------------------------------------------
// Embedding gather + sinusoidal position embedding (positions start at 1)
// ---------------------------------------------------------------------------
__global__ __launch_bounds__(256)
void embed_pos(const int* __restrict__ ids, const float* __restrict__ emb,
               float* __restrict__ out)
{
  const int s  = blockIdx.x;
  const int n  = s & 255;
  const int id = ids[s];
  const float p = (float)(n + 1);
#pragma unroll
  for (int rep = 0; rep < 2; ++rep) {
    int e  = threadIdx.x + rep * 256;
    int j2 = e & ~1;  // 2j
    float ang = p * __expf((float)j2 * (-9.21034037f / 512.f)); // ln(1e4)
    float val = (e & 1) ? __cosf(ang) : __sinf(ang);
    out[(size_t)s * 512 + e] = emb[(size_t)id * 512 + e] + val;
  }
}

// ---------------------------------------------------------------------------
// Orchestration
// d_in layout (jax pytree order; params dict flattened with sorted keys):
//  0 product_ids  1 reactants_ids  2 product_mask  3 reactants_mask
//  4 mutual_mask  5.. dec[0..3] (22 leaves each), emb_src, emb_tgt,
//  enc[0..3] (12 leaves each)
// dec leaves: W1 W2 Wk Wo Wo2 Wq Wqkv Wv b1 b2 bk bo bo2 bq bqkv bv
//             ln1_b ln1_g ln2_b ln2_g ln3_b ln3_g
// enc leaves: W1 W2 Wo Wqkv b1 b2 bo bqkv ln1_b ln1_g ln2_b ln2_g
// ---------------------------------------------------------------------------
extern "C" void kernel_launch(void* const* d_in, const int* in_sizes, int n_in,
                              void* d_out, int out_size, void* d_ws, size_t ws_size,
                              hipStream_t stream)
{
  (void)in_sizes; (void)n_in; (void)out_size; (void)ws_size;
  constexpr int M = 4096;               // B*L*256 token rows
  const int*   prod_ids = (const int*)d_in[0];
  const int*   reac_ids = (const int*)d_in[1];
  const float* prod_mask = (const float*)d_in[2];
  const float* reac_mask = (const float*)d_in[3];
  const float* mut_mask  = (const float*)d_in[4];
  auto P = [&](int i) { return (const float*)d_in[i]; };
  const int DEC0 = 5, EMB_SRC = 93, EMB_TGT = 94, ENC0 = 95;

  float* ws      = (float*)d_ws;
  float* bufX    = ws;                           // [4096,512]
  float* bufY    = bufX   + (size_t)M * 512;     // [4096,512]
  float* bufA    = bufY   + (size_t)M * 512;     // [4096,512] merged attn
  float* bufENC  = bufA   + (size_t)M * 512;     // [4096,512] encoder out
  float* bufQKV  = bufENC + (size_t)M * 512;     // [4096,1536]
  float* bufH    = bufQKV + (size_t)M * 1536;    // [4096,2048]
  float* bufS    = bufH;                         // scores alias H (disjoint)
  __bf16* bufP   = (__bf16*)(bufH + (size_t)M * 2048); // [128,256,256] bf16

  auto gemm = [&](const float* A, int lda, const float* W, int ldw,
                  const float* bias, const float* res, int ldres,
                  float* C, int ldc, int N, int K, int relu) {
    dim3 grid(N / 128, M / 128);
    gemm_bf16_wmma<<<grid, 256, 0, stream>>>(A, lda, W, ldw, bias,
                                             res, ldres, C, ldc, K, relu);
  };
  auto attention = [&](const float* mask) {
    attn_scores_wmma<<<dim3(2, 2, 128), 256, 0, stream>>>(bufQKV, mask, bufS, 0, 512);
    softmax256_bf16<<<dim3(128 * 256), 128, 0, stream>>>(bufS, bufP);
    attn_out_wmma<<<dim3(2, 128), 256, 0, stream>>>(bufP, bufQKV, bufA, 1024);
  };
  auto lnorm = [&](const float* X, const float* g, const float* b, float* Y) {
    layernorm512<<<dim3(M), 256, 0, stream>>>(X, g, b, Y);
  };

  // ---------------- encoder ----------------
  embed_pos<<<dim3(M), 256, 0, stream>>>(prod_ids, P(EMB_SRC), bufX);
  for (int l = 0; l < 4; ++l) {
    int e = ENC0 + 12 * l;
    const float *W1 = P(e+0), *W2 = P(e+1), *Wo = P(e+2), *Wqkv = P(e+3);
    const float *b1 = P(e+4), *b2 = P(e+5), *bo = P(e+6), *bqkv = P(e+7);
    const float *ln1b = P(e+8), *ln1g = P(e+9), *ln2b = P(e+10), *ln2g = P(e+11);
    gemm(bufX, 512, Wqkv, 1536, bqkv, nullptr, 0, bufQKV, 1536, 1536, 512, 0);
    attention(prod_mask);
    gemm(bufA, 512, Wo, 512, bo, bufX, 512, bufY, 512, 512, 512, 0);
    lnorm(bufY, ln1g, ln1b, bufX);
    gemm(bufX, 512, W1, 2048, b1, nullptr, 0, bufH, 2048, 2048, 512, 1);
    gemm(bufH, 2048, W2, 512, b2, bufX, 512, bufY, 512, 512, 2048, 0);
    lnorm(bufY, ln2g, ln2b, bufX);
  }
  (void)hipMemcpyAsync(bufENC, bufX, (size_t)M * 512 * sizeof(float),
                       hipMemcpyDeviceToDevice, stream);

  // ---------------- decoder ----------------
  embed_pos<<<dim3(M), 256, 0, stream>>>(reac_ids, P(EMB_TGT), bufX);
  for (int l = 0; l < 4; ++l) {
    int d0 = DEC0 + 22 * l;
    const float *W1 = P(d0+0), *W2 = P(d0+1), *Wk = P(d0+2), *Wo = P(d0+3);
    const float *Wo2 = P(d0+4), *Wq = P(d0+5), *Wqkv = P(d0+6), *Wv = P(d0+7);
    const float *b1 = P(d0+8), *b2 = P(d0+9), *bk = P(d0+10), *bo = P(d0+11);
    const float *bo2 = P(d0+12), *bq = P(d0+13), *bqkv = P(d0+14), *bv = P(d0+15);
    const float *ln1b = P(d0+16), *ln1g = P(d0+17);
    const float *ln2b = P(d0+18), *ln2g = P(d0+19);
    const float *ln3b = P(d0+20), *ln3g = P(d0+21);
    // self-attention
    gemm(bufX, 512, Wqkv, 1536, bqkv, nullptr, 0, bufQKV, 1536, 1536, 512, 0);
    attention(reac_mask);
    gemm(bufA, 512, Wo, 512, bo, bufX, 512, bufY, 512, 512, 512, 0);
    lnorm(bufY, ln1g, ln1b, bufX);
    // cross-attention: q from x, k/v from encoder output, packed into QKV cols
    gemm(bufX,   512, Wq, 512, bq, nullptr, 0, bufQKV + 0,    1536, 512, 512, 0);
    gemm(bufENC, 512, Wk, 512, bk, nullptr, 0, bufQKV + 512,  1536, 512, 512, 0);
    gemm(bufENC, 512, Wv, 512, bv, nullptr, 0, bufQKV + 1024, 1536, 512, 512, 0);
    attention(mut_mask);
    gemm(bufA, 512, Wo2, 512, bo2, bufX, 512, bufY, 512, 512, 512, 0);
    lnorm(bufY, ln2g, ln2b, bufX);
    // FFN
    gemm(bufX, 512, W1, 2048, b1, nullptr, 0, bufH, 2048, 2048, 512, 1);
    gemm(bufH, 2048, W2, 512, b2, bufX, 512, bufY, 512, 512, 2048, 0);
    lnorm(bufY, ln3g, ln3b, bufX);
  }
  (void)hipMemcpyAsync(d_out, bufX, (size_t)M * 512 * sizeof(float),
                       hipMemcpyDeviceToDevice, stream);
}
